// AllConv_63660005261511
// MI455X (gfx1250) — matrix operand order repaired and checked by reading, not compiled, not run on hardware
//
#include <hip/hip_runtime.h>

// ---------------------------------------------------------------- types
typedef __attribute__((ext_vector_type(16))) _Float16 v16h;
typedef __attribute__((ext_vector_type(8)))  _Float16 v8h;
typedef __attribute__((ext_vector_type(8)))  float    v8f;

union AFrag { v16h v; v8h h[2]; };

// ---------------------------------------------------------------- sizes
#define NN      50000
#define EE      1600000
#define IN_NFE  64
#define IN_EFE  16
#define HH      12
#define WPB     4            // waves per block (128 threads)

// f16 weight-fragment offsets (in halves) inside workspace
#define MSG_L0  0            // Kp=160 (5 ktiles) x 4 ntiles x 512
#define MSG_L1  10240        // 2x4x512
#define MSG_L2  14336
#define MSG_L3  18432        // 2x4x512 (N padded 49->64)
#define RED_L0  22528        // Kp=128 (4 ktiles) x 4 x 512
#define RED_L1  30720
#define RED_L2  34816
#define RED_L3  38912
#define W_HALVES 43008

// ------------------------------------------------- ordered float <-> uint
__device__ __forceinline__ unsigned f2ord(float f) {
    unsigned u = __float_as_uint(f);
    return (u & 0x80000000u) ? ~u : (u | 0x80000000u);
}
__device__ __forceinline__ float ord2f(unsigned u) {
    return (u & 0x80000000u) ? __uint_as_float(u & 0x7FFFFFFFu)
                             : __uint_as_float(~u);
}

// ------------------------------------------------- weight swizzle kernel
// Store W (Kreal x Nreal, row-major) as WMMA-B fragments:
// frag[tile][lane][h] with K = kt*32 + (h&7) + (h&8?16:0) + (lane<16?0:8), N = nt*16 + (lane&15)
__global__ void prep_weights(const float* __restrict__ W, int Kreal, int Nreal,
                             int Kp, _Float16* __restrict__ dst) {
    int total = (Kp / 32) * 4 * 512;
    for (int idx = blockIdx.x * blockDim.x + threadIdx.x; idx < total;
         idx += gridDim.x * blockDim.x) {
        int tile = idx >> 9;
        int rem  = idx & 511;
        int lane = rem >> 4;
        int h    = rem & 15;
        int kt   = tile >> 2;        // nTiles == 4 always (Np == 64)
        int nt   = tile & 3;
        int K    = kt * 32 + (h & 7) + ((h & 8) ? 16 : 0) + ((lane < 16) ? 0 : 8);
        int Ncol = nt * 16 + (lane & 15);
        float v  = (K < Kreal && Ncol < Nreal) ? W[K * Nreal + Ncol] : 0.0f;
        dst[idx] = (_Float16)v;
    }
}

// ------------------------------------------------- accumulator init
__global__ void init_acc(float* a1, float* a2, unsigned* a3, unsigned* a4,
                         unsigned* cnt) {
    int i = blockIdx.x * blockDim.x + threadIdx.x;
    int M = NN * HH;
    if (i < M) { a1[i] = 0.0f; a2[i] = 0.0f; a3[i] = 0xFFFFFFFFu; a4[i] = 0u; }
    if (i < NN) cnt[i] = 0u;
}

// ------------------------------------------------- WMMA GEMM helpers
// One wave computes a 16(row) x 64(col) layer: A from LDS (row-major f16),
// B fragments from pre-swizzled weights. All LDS traffic is wave-private and
// LDS ops are in-order per wave, so no block barriers are required.
template <int KT>
__device__ __forceinline__ void gemm16(const _Float16* __restrict__ inRow,
                                       const _Float16* __restrict__ wFrag,
                                       int lane, v8f acc[4]) {
    const int base = (lane & 16) ? 8 : 0;
#pragma unroll
    for (int kt = 0; kt < KT; ++kt) {
        AFrag a;
        a.h[0] = *(const v8h*)(inRow + kt * 32 + base);
        a.h[1] = *(const v8h*)(inRow + kt * 32 + 16 + base);
#pragma unroll
        for (int nt = 0; nt < 4; ++nt) {
            v16h b = *(const v16h*)(wFrag + (size_t)(kt * 4 + nt) * 512 + lane * 16);
            acc[nt] = __builtin_amdgcn_wmma_f32_16x16x32_f16(
                false, a.v, false, b, (short)0, acc[nt], false, false);
        }
    }
}

// D-layout (lane = col, VGPR = row) -> row-major f16 activation buffer
__device__ __forceinline__ void bias_relu_store(const v8f acc[4],
                                                const float* __restrict__ bias,
                                                _Float16* __restrict__ act,
                                                int lane) {
    const int col0 = lane & 15;
    const int rb   = (lane & 16) ? 8 : 0;
#pragma unroll
    for (int nt = 0; nt < 4; ++nt) {
        int n  = nt * 16 + col0;
        float b = bias[n];
#pragma unroll
        for (int i = 0; i < 8; ++i) {
            float v = fmaxf(acc[nt][i] + b, 0.0f);
            act[(rb + i) * 64 + n] = (_Float16)v;
        }
    }
}

#define ZERO_ACC(acc) { v8f z_[4] = {}; acc[0]=z_[0];acc[1]=z_[1];acc[2]=z_[2];acc[3]=z_[3]; }

// ------------------------------------------------- edge (message) kernel
__global__ __launch_bounds__(128) void edge_kernel(
    const float* __restrict__ nf, const float* __restrict__ ef,
    const int* __restrict__ src, const int* __restrict__ dst,
    const _Float16* __restrict__ wH,
    const float* __restrict__ b0, const float* __restrict__ b1,
    const float* __restrict__ b2, const float* __restrict__ b3,
    float* acc1, float* acc2, unsigned* acc3, unsigned* acc4, unsigned* cnt) {
    __shared__ _Float16 sIn[WPB][16 * 160];
    __shared__ _Float16 sAct[WPB][16 * 64];
    __shared__ float    sXout[WPB][16 * 64];

    const int wave = threadIdx.x >> 5;
    const int lane = threadIdx.x & 31;
    const int numTiles = EE / 16;
    int tileRaw = blockIdx.x * WPB + wave;
    bool active = tileRaw < numTiles;
    int tile = active ? tileRaw : (numTiles - 1);

    const int r  = lane & 15;
    const int hf = lane >> 4;
    const int e  = tile * 16 + r;
    const int s  = src[e];
    const int d  = dst[e];

    if (threadIdx.x == 0) __builtin_prefetch(wH, 0, 1);   // global_prefetch_b8

    _Float16* in = &sIn[wave][0];
    // stage [16 x 160] f16: cols 0-63 = nf[src], 64-127 = nf[dst],
    // 128-143 = ef, 144-159 = zero pad
    {
        const float* np_ = nf + (size_t)(hf ? d : s) * IN_NFE;
#pragma unroll
        for (int c = 0; c < IN_NFE; c += 2) {
            float2 f = *(const float2*)(np_ + c);
            in[r * 160 + hf * 64 + c]     = (_Float16)f.x;
            in[r * 160 + hf * 64 + c + 1] = (_Float16)f.y;
        }
        if (hf == 0) {
            const float* ep = ef + (size_t)e * IN_EFE;
#pragma unroll
            for (int c = 0; c < IN_EFE; c += 2) {
                float2 f = *(const float2*)(ep + c);
                in[r * 160 + 128 + c]     = (_Float16)f.x;
                in[r * 160 + 128 + c + 1] = (_Float16)f.y;
            }
#pragma unroll
            for (int c = 0; c < 16; ++c) in[r * 160 + 144 + c] = (_Float16)0.0f;
        }
    }

    _Float16* act = &sAct[wave][0];
    v8f acc[4];

    // L0: 160x64
    ZERO_ACC(acc);
    gemm16<5>(in + r * 160, wH + MSG_L0, lane, acc);
    bias_relu_store(acc, b0, act, lane);
    // L1
    ZERO_ACC(acc);
    gemm16<2>(act + r * 64, wH + MSG_L1, lane, acc);
    bias_relu_store(acc, b1, act, lane);
    // L2
    ZERO_ACC(acc);
    gemm16<2>(act + r * 64, wH + MSG_L2, lane, acc);
    bias_relu_store(acc, b2, act, lane);
    // L3 (linear, Nreal = 49) -> f32 LDS buffer
    ZERO_ACC(acc);
    gemm16<2>(act + r * 64, wH + MSG_L3, lane, acc);
    float* xout = &sXout[wave][0];
    {
        const int col0 = lane & 15;
        const int rb   = (lane & 16) ? 8 : 0;
#pragma unroll
        for (int nt = 0; nt < 4; ++nt) {
            int n  = nt * 16 + col0;
            float b = (n < 49) ? b3[n] : 0.0f;
#pragma unroll
            for (int i = 0; i < 8; ++i) xout[(rb + i) * 64 + n] = acc[nt][i] + b;
        }
    }

    // epilogue: gate + scatter reductions (LDS in-order per wave -> no barrier)
    float gate = 1.0f / (1.0f + expf(-xout[r * 64 + 0]));
    if (active) {
        if (hf == 0) {
            atomicAdd(&cnt[d], 1u);
#pragma unroll
            for (int j = 0; j < HH; ++j) {
                atomicAdd(&acc1[d * HH + j], xout[r * 64 + 1 + j] * gate);
                atomicAdd(&acc2[d * HH + j], xout[r * 64 + 13 + j] * gate);
            }
        } else {
#pragma unroll
            for (int j = 0; j < HH; ++j) {
                atomicMin(&acc3[d * HH + j], f2ord(xout[r * 64 + 25 + j] * gate));
                atomicMax(&acc4[d * HH + j], f2ord(xout[r * 64 + 37 + j] * gate));
            }
        }
    }
}

// ------------------------------------------------- node (reduce) kernel
__global__ __launch_bounds__(128) void node_kernel(
    const float* __restrict__ nf, const _Float16* __restrict__ wH,
    const float* __restrict__ b0, const float* __restrict__ b1,
    const float* __restrict__ b2, const float* __restrict__ b3,
    const float* __restrict__ acc1, const float* __restrict__ acc2,
    const unsigned* __restrict__ acc3, const unsigned* __restrict__ acc4,
    const unsigned* __restrict__ cnt, float* __restrict__ out) {
    __shared__ _Float16 sIn[WPB][16 * 128];
    __shared__ _Float16 sAct[WPB][16 * 64];

    const int wave = threadIdx.x >> 5;
    const int lane = threadIdx.x & 31;
    const int numTiles = NN / 16;      // 3125
    int tileRaw = blockIdx.x * WPB + wave;
    bool active = tileRaw < numTiles;
    int tile = active ? tileRaw : (numTiles - 1);

    const int r    = lane & 15;
    const int hf   = lane >> 4;
    const int node = tile * 16 + r;

    _Float16* in = &sIn[wave][0];
    if (hf == 0) {
        const float* np_ = nf + (size_t)node * IN_NFE;
#pragma unroll
        for (int c = 0; c < IN_NFE; c += 2) {
            float2 f = *(const float2*)(np_ + c);
            in[r * 128 + c]     = (_Float16)f.x;
            in[r * 128 + c + 1] = (_Float16)f.y;
        }
    } else {
        float cn  = (float)cnt[node];
        float inv = 1.0f / fmaxf(cn, 1.0f);
        bool has  = cn > 0.0f;
#pragma unroll
        for (int j = 0; j < HH; ++j) {
            in[r * 128 + 64 + j]  = (_Float16)(acc1[node * HH + j] * inv);
            in[r * 128 + 76 + j]  = (_Float16)(acc2[node * HH + j]);
            in[r * 128 + 88 + j]  = (_Float16)(has ? ord2f(acc3[node * HH + j]) : 0.0f);
            in[r * 128 + 100 + j] = (_Float16)(has ? ord2f(acc4[node * HH + j]) : 0.0f);
        }
#pragma unroll
        for (int c = 112; c < 128; ++c) in[r * 128 + c] = (_Float16)0.0f;
    }

    _Float16* act = &sAct[wave][0];
    v8f acc[4];

    ZERO_ACC(acc);
    gemm16<4>(in + r * 128, wH + RED_L0, lane, acc);
    bias_relu_store(acc, b0, act, lane);
    ZERO_ACC(acc);
    gemm16<2>(act + r * 64, wH + RED_L1, lane, acc);
    bias_relu_store(acc, b1, act, lane);
    ZERO_ACC(acc);
    gemm16<2>(act + r * 64, wH + RED_L2, lane, acc);
    bias_relu_store(acc, b2, act, lane);
    ZERO_ACC(acc);
    gemm16<2>(act + r * 64, wH + RED_L3, lane, acc);

    if (active) {
        const int col0 = lane & 15;
        const int rb   = (lane & 16) ? 8 : 0;
#pragma unroll
        for (int nt = 0; nt < 4; ++nt) {
            int n = nt * 16 + col0;
            float b = b3[n];
#pragma unroll
            for (int i = 0; i < 8; ++i)
                out[(size_t)(tile * 16 + rb + i) * 64 + n] = acc[nt][i] + b;
        }
    }
}

// ------------------------------------------------- launcher
extern "C" void kernel_launch(void* const* d_in, const int* in_sizes, int n_in,
                              void* d_out, int out_size, void* d_ws, size_t ws_size,
                              hipStream_t stream) {
    (void)in_sizes; (void)n_in; (void)out_size; (void)ws_size;
    const float* nf  = (const float*)d_in[0];
    const float* ef  = (const float*)d_in[1];
    const int*   src = (const int*)d_in[2];
    const int*   dst = (const int*)d_in[3];
    const float* msgW[4] = {(const float*)d_in[4],  (const float*)d_in[5],
                            (const float*)d_in[6],  (const float*)d_in[7]};
    const float* msgB[4] = {(const float*)d_in[8],  (const float*)d_in[9],
                            (const float*)d_in[10], (const float*)d_in[11]};
    const float* redW[4] = {(const float*)d_in[12], (const float*)d_in[13],
                            (const float*)d_in[14], (const float*)d_in[15]};
    const float* redB[4] = {(const float*)d_in[16], (const float*)d_in[17],
                            (const float*)d_in[18], (const float*)d_in[19]};
    float* out = (float*)d_out;

    char* ws = (char*)d_ws;
    _Float16* wH = (_Float16*)ws;
    size_t off = (size_t)W_HALVES * 2;                       // 86016 B
    float*    acc1 = (float*)(ws + off);    off += (size_t)NN * HH * 4;
    float*    acc2 = (float*)(ws + off);    off += (size_t)NN * HH * 4;
    unsigned* acc3 = (unsigned*)(ws + off); off += (size_t)NN * HH * 4;
    unsigned* acc4 = (unsigned*)(ws + off); off += (size_t)NN * HH * 4;
    unsigned* cnt  = (unsigned*)(ws + off); off += (size_t)NN * 4;

    // swizzle weights to f16 B-fragments
    const int   Kr[8]  = {144, 64, 64, 64, 112, 64, 64, 64};
    const int   Nr[8]  = {64, 64, 64, 49, 64, 64, 64, 64};
    const int   Kp[8]  = {160, 64, 64, 64, 128, 64, 64, 64};
    const int   oH[8]  = {MSG_L0, MSG_L1, MSG_L2, MSG_L3,
                          RED_L0, RED_L1, RED_L2, RED_L3};
    const float* Wp[8] = {msgW[0], msgW[1], msgW[2], msgW[3],
                          redW[0], redW[1], redW[2], redW[3]};
    for (int l = 0; l < 8; ++l) {
        int total = (Kp[l] / 32) * 4 * 512;
        prep_weights<<<(total + 255) / 256, 256, 0, stream>>>(
            Wp[l], Kr[l], Nr[l], Kp[l], wH + oH[l]);
    }

    init_acc<<<(NN * HH + 255) / 256, 256, 0, stream>>>(acc1, acc2, acc3, acc4, cnt);

    const int edgeTiles = EE / 16;                 // 100000
    edge_kernel<<<edgeTiles / WPB, 128, 0, stream>>>(
        nf, ef, src, dst, wH, msgB[0], msgB[1], msgB[2], msgB[3],
        acc1, acc2, acc3, acc4, cnt);

    const int nodeTiles = NN / 16;                 // 3125
    node_kernel<<<(nodeTiles + WPB - 1) / WPB, 128, 0, stream>>>(
        nf, wH, redB[0], redB[1], redB[2], redB[3],
        acc1, acc2, acc3, acc4, cnt, out);
}
